// LSTM_15556371546645
// MI455X (gfx1250) — matrix operand order repaired and checked by reading, not compiled, not run on hardware
//
#include <hip/hip_runtime.h>

// ---------------------------------------------------------------------------
// CDNA5 (gfx1250) LSTM: bf16 WMMA (v_wmma_f32_16x16x32_bf16), wave32.
// Persistent-workgroup recurrence with LDS-resident weights and a global
// split barrier per timestep. Encoder/decoder are row-looping WMMA GEMMs
// with fragment-native bf16 weights staged once per block in LDS.
// ---------------------------------------------------------------------------

typedef __attribute__((ext_vector_type(16))) __bf16 v16bf;
typedef __attribute__((ext_vector_type(8)))  float  v8f;

union Frag {
  v16bf  v;
  uint4  q[2];
  __bf16 h[16];
};

// Native converts / transcendentals where gfx1250 exposes them.
__device__ __forceinline__ float fexp(float x) {
#if __has_builtin(__builtin_amdgcn_exp2f)
  return __builtin_amdgcn_exp2f(x * 1.44269504088896341f);   // v_exp_f32
#else
  return __expf(x);
#endif
}
__device__ __forceinline__ float frcp(float x) {
#if __has_builtin(__builtin_amdgcn_rcpf)
  return __builtin_amdgcn_rcpf(x);                           // v_rcp_f32
#else
  return 1.0f / x;
#endif
}
__device__ __forceinline__ float sigm(float x) { return frcp(1.0f + fexp(-x)); }
__device__ __forceinline__ float tanh_fast(float x) {
#if __has_builtin(__builtin_amdgcn_tanhf)
  return __builtin_amdgcn_tanhf(x);                          // v_tanh_f32
#else
  return 2.0f * sigm(2.0f * x) - 1.0f;
#endif
}

// ---------------------------------------------------------------------------
// Encoder: x0[r][n] = sum_k in[r][k] * enc_w[n][k] + enc_b[n]   (bf16 out)
// r in [0,131072), n in [0,256), k in [0,64).
// Grid 1024 x 256thr. Block stages enc_w (bf16, frag-native, 32 KiB) once,
// then loops 8 M-tiles; wave w covers N-tiles {w, w+8}.
// ---------------------------------------------------------------------------
__global__ __launch_bounds__(256) void enc_kernel(
    const float* __restrict__ in, const float* __restrict__ ew,
    const float* __restrict__ eb, __bf16* __restrict__ x0)
{
  __shared__ __bf16 blds[32 * 32 * 16];        // [nt(16)][kt(2)][lane(32)][16]
  const int tid = threadIdx.x;
  const int w = tid >> 5, lane = tid & 31;
  const int lrow = lane & 15, lhalf = lane >> 4;

  // stage enc_w -> LDS (fragment-native): n = nt*16+lrow, k = kt*32+half*16+e
  for (int tile = w; tile < 32; tile += 8) {
    const int nt = tile >> 1, kt = tile & 1;
    const float* src = ew + (size_t)(nt * 16 + lrow) * 64 + kt * 32 + lhalf * 16;
    Frag f;
    for (int e = 0; e < 16; ++e) f.h[e] = (__bf16)src[e];
    uint4* dst = (uint4*)(blds + ((size_t)tile * 32 + lane) * 16);
    dst[0] = f.q[0]; dst[1] = f.q[1];
  }
  __syncthreads();

  const int nt[2] = { w, w + 8 };
  const float bias0 = eb[nt[0] * 16 + lrow];
  const float bias1 = eb[nt[1] * 16 + lrow];

  for (int it = 0; it < 8; ++it) {
    const size_t mbase = ((size_t)blockIdx.x * 8 + it) * 16;
    v8f acc[2];
    for (int r = 0; r < 8; ++r) { acc[0][r] = bias0; acc[1][r] = bias1; }

    for (int kt = 0; kt < 2; ++kt) {
      Frag a;                 // A 16x32: lane = row, k = half*16 + {0..15}
      const float* ap = in + (mbase + lrow) * 64 + kt * 32 + lhalf * 16;
      for (int e = 0; e < 8; ++e) a.h[e]     = (__bf16)ap[e];
      for (int e = 0; e < 8; ++e) a.h[8 + e] = (__bf16)ap[8 + e];
      for (int i = 0; i < 2; ++i) {
        const uint4* bp = (const uint4*)(blds + (((size_t)nt[i] * 2 + kt) * 32 + lane) * 16);
        Frag b; b.q[0] = bp[0]; b.q[1] = bp[1];
        acc[i] = __builtin_amdgcn_wmma_f32_16x16x32_bf16(
                     false, a.v, false, b.v, (short)0, acc[i], false, false);
      }
    }
    for (int i = 0; i < 2; ++i)
      for (int r = 0; r < 8; ++r) {
        size_t m = mbase + lhalf * 8 + r;
        x0[m * 256 + (size_t)nt[i] * 16 + lrow] = (__bf16)acc[i][r];
      }
  }
}

// ---------------------------------------------------------------------------
// LSTM layer: persistent kernel, grid = 8 blocks x 256 thr (8 waves).
// Block `wg` owns h columns [wg*32, wg*32+32) -> 128 gate rows (i/f/g/o x 32).
// Weight slice (128 x 512, K = [x|h]) staged bf16 in LDS, fragment-native:
//   wlds[nt(8)][kt(16)][lane(32)][16]  = 128 KiB (dynamic LDS, WGP has 320 KiB).
// Wave w: mt = w>>1 (batch tile), jn = w&1; acc tiles nt = jn + 2*gate ->
// i/f/g/o for one (m,n) land in the same lane => register-resident cell state.
// Cross-WG h exchange through global x_out + release/acquire split barrier.
// ---------------------------------------------------------------------------
__global__ __launch_bounds__(256) void lstm_kernel(
    const __bf16* __restrict__ xin, __bf16* __restrict__ xout,
    const float* __restrict__ wih, const float* __restrict__ whh,
    const float* __restrict__ bih, const float* __restrict__ bhh,
    unsigned int* __restrict__ cnt)
{
  extern __shared__ __bf16 wlds[];             // [8][16][32][16] bf16
  const int wg = blockIdx.x;
  const int tid = threadIdx.x;
  const int w = tid >> 5, lane = tid & 31;
  const int lrow = lane & 15, lhalf = lane >> 4;

  // ---- stage weight slice (fp32 -> bf16) fragment-native into LDS ----
  for (int tile = w; tile < 128; tile += 8) {
    const int nt = tile >> 4, kt = tile & 15;
    const int gate = nt >> 1, jn2 = nt & 1;
    const int grow = gate * 256 + wg * 32 + jn2 * 16 + lrow;   // gate row 0..1023
    const int kb = kt * 32 + lhalf * 16;                       // B frag: k = half*16+e
    Frag f;
    for (int e = 0; e < 16; ++e) {
      int k = kb + e;
      float v = (k < 256) ? wih[(size_t)grow * 256 + k]
                          : whh[(size_t)grow * 256 + (k - 256)];
      f.h[e] = (__bf16)v;
    }
    uint4* dst = (uint4*)(wlds + ((size_t)tile * 32 + lane) * 16);
    dst[0] = f.q[0]; dst[1] = f.q[1];
  }
  __syncthreads();

  const int mt = w >> 1, jn = w & 1;
  float bias[4];
  for (int g = 0; g < 4; ++g) {
    int grow = g * 256 + wg * 32 + jn * 16 + lrow;
    bias[g] = bih[grow] + bhh[grow];
  }
  float c[8];
  for (int r = 0; r < 8; ++r) c[r] = 0.0f;

  const int    mrow = mt * 16 + lrow;                    // batch row for A loads
  const size_t ncol = (size_t)wg * 32 + jn * 16 + lrow;  // h column for output

  for (int t = 0; t < 2048; ++t) {
    v8f acc[4];
    for (int g = 0; g < 4; ++g)
      for (int r = 0; r < 8; ++r) acc[g][r] = bias[g];

    const __bf16* xr = xin  + ((size_t)t * 64 + mrow) * 256;
    const __bf16* hr = xout + ((size_t)(t - 1) * 64 + mrow) * 256;
    if (t + 1 < 2048)   // global_prefetch of next timestep's x row
      __builtin_prefetch(xin + ((size_t)(t + 1) * 64 + mrow) * 256, 0, 1);

    const int ktend = (t == 0) ? 8 : 16;       // h == 0 at t=0: skip h K-tiles
    for (int kt = 0; kt < ktend; ++kt) {
      Frag a;                                  // A: two 16 B runs per lane
      const __bf16* src = (kt < 8) ? xr : hr;
      const int kb = (kt & 7) * 32 + lhalf * 16;
      a.q[0] = *(const uint4*)(src + kb);
      a.q[1] = *(const uint4*)(src + kb + 8);
      for (int g = 0; g < 4; ++g) {
        const int nt = jn + 2 * g;
        const uint4* bp =
            (const uint4*)(wlds + (((size_t)nt * 16 + kt) * 32 + lane) * 16);
        Frag b; b.q[0] = bp[0]; b.q[1] = bp[1];
        acc[g] = __builtin_amdgcn_wmma_f32_16x16x32_bf16(
                     false, a.v, false, b.v, (short)0, acc[g], false, false);
      }
    }

    // gates -> cell/hidden update in registers; scatter h (bf16) to global
    for (int r = 0; r < 8; ++r) {
      float iv = sigm(acc[0][r]);
      float fv = sigm(acc[1][r]);
      float gv = tanh_fast(acc[2][r]);
      float ov = sigm(acc[3][r]);
      c[r] = fv * c[r] + iv * gv;
      float hv = ov * tanh_fast(c[r]);
      size_t m = (size_t)mt * 16 + lhalf * 8 + r;
      xout[((size_t)t * 64 + m) * 256 + ncol] = (__bf16)hv;
    }

    // ---- inter-workgroup split barrier (8 arrivals per step) ----
    __threadfence();
    __syncthreads();
    if (tid == 0) {
      __hip_atomic_fetch_add(cnt, 1u, __ATOMIC_RELEASE, __HIP_MEMORY_SCOPE_AGENT);
      const unsigned int target = 8u * (unsigned int)(t + 1);
      while (__hip_atomic_load(cnt, __ATOMIC_ACQUIRE, __HIP_MEMORY_SCOPE_AGENT) < target)
        __builtin_amdgcn_s_sleep(1);
    }
    __syncthreads();
    __threadfence();
  }
}

// ---------------------------------------------------------------------------
// Decoder: out[r][n] = sum_k x2[r][k] * dec_w[n][k] + dec_b[n]   (fp32 out)
// n in [0,64). Grid 512 x 256thr. Block stages dec_w (bf16, frag-native,
// 32 KiB) once, then loops 8 iterations; waves cover 2 M-tiles x 4 N-tiles.
// ---------------------------------------------------------------------------
__global__ __launch_bounds__(256) void dec_kernel(
    const __bf16* __restrict__ x2, const float* __restrict__ dw,
    const float* __restrict__ db, float* __restrict__ out)
{
  __shared__ __bf16 blds[32 * 32 * 16];        // [nt(4)][kt(8)][lane(32)][16]
  const int tid = threadIdx.x;
  const int w = tid >> 5, lane = tid & 31;
  const int lrow = lane & 15, lhalf = lane >> 4;

  // stage dec_w -> LDS (fragment-native): n = nt*16+lrow, k = kt*32+half*16+e
  for (int tile = w; tile < 32; tile += 8) {
    const int nt = tile >> 3, kt = tile & 7;
    const float* src = dw + (size_t)(nt * 16 + lrow) * 256 + kt * 32 + lhalf * 16;
    Frag f;
    for (int e = 0; e < 16; ++e) f.h[e] = (__bf16)src[e];
    uint4* dst = (uint4*)(blds + ((size_t)tile * 32 + lane) * 16);
    dst[0] = f.q[0]; dst[1] = f.q[1];
  }
  __syncthreads();

  const int nt = w & 3;
  const int ncol = nt * 16 + lrow;
  const float bias = db[ncol];

  for (int it = 0; it < 8; ++it) {
    const size_t rowt = ((size_t)blockIdx.x * 8 + it) * 2 + (w >> 2);
    const size_t mrow = rowt * 16 + lrow;

    v8f acc;
    for (int r = 0; r < 8; ++r) acc[r] = bias;
    for (int kt = 0; kt < 8; ++kt) {
      Frag a;
      const __bf16* ap = x2 + mrow * 256 + kt * 32 + lhalf * 16;
      a.q[0] = *(const uint4*)(ap);
      a.q[1] = *(const uint4*)(ap + 8);
      const uint4* bp = (const uint4*)(blds + (((size_t)nt * 8 + kt) * 32 + lane) * 16);
      Frag b; b.q[0] = bp[0]; b.q[1] = bp[1];
      acc = __builtin_amdgcn_wmma_f32_16x16x32_bf16(
                false, a.v, false, b.v, (short)0, acc, false, false);
    }
    for (int r = 0; r < 8; ++r) {
      size_t m = rowt * 16 + lhalf * 8 + r;
      out[m * 64 + ncol] = acc[r];
    }
  }
}

// ---------------------------------------------------------------------------
extern "C" void kernel_launch(void* const* d_in, const int* in_sizes, int n_in,
                              void* d_out, int out_size, void* d_ws, size_t ws_size,
                              hipStream_t stream) {
  (void)in_sizes; (void)n_in; (void)out_size; (void)ws_size;
  const float* in_ids = (const float*)d_in[0];   // (2048, 64, 64)
  const float* enc_w  = (const float*)d_in[1];   // (256, 64)
  const float* enc_b  = (const float*)d_in[2];   // (256,)
  const float* w_ih   = (const float*)d_in[3];   // (2, 1024, 256)
  const float* w_hh   = (const float*)d_in[4];   // (2, 1024, 256)
  const float* b_ih   = (const float*)d_in[5];   // (2, 1024)
  const float* b_hh   = (const float*)d_in[6];   // (2, 1024)
  const float* dec_w  = (const float*)d_in[7];   // (64, 256)
  const float* dec_b  = (const float*)d_in[8];   // (64,)
  float* out = (float*)d_out;                    // (2048, 64, 64) fp32

  char* ws = (char*)d_ws;
  unsigned int* counters = (unsigned int*)ws;                 // 2 x u32
  const size_t  XBYTES   = (size_t)131072 * 256 * 2;          // 64 MiB
  __bf16* x0 = (__bf16*)(ws + 256);
  __bf16* x1 = (__bf16*)(ws + 256 + XBYTES);

  hipMemsetAsync(ws, 0, 256, stream);   // reset barrier counters (graph-safe)

  enc_kernel<<<1024, 256, 0, stream>>>(in_ids, enc_w, enc_b, x0);
  lstm_kernel<<<8, 256, 131072, stream>>>(x0, x1, w_ih, w_hh, b_ih, b_hh,
                                          counters + 0);
  lstm_kernel<<<8, 256, 131072, stream>>>(x1, x0,             // x2 aliases x0
                                          w_ih + (size_t)1024 * 256,
                                          w_hh + (size_t)1024 * 256,
                                          b_ih + 1024, b_hh + 1024,
                                          counters + 1);
  dec_kernel<<<512, 256, 0, stream>>>(x0, dec_w, dec_b, out);
}
